// WindowAttention3D_37314675868161
// MI455X (gfx1250) — compile-verified, matrix operand order checked by reading, not run on hardware
//
#include <hip/hip_runtime.h>
#include <hip/hip_bf16.h>
#include <cstdint>

typedef __attribute__((ext_vector_type(16))) __bf16       v16bf;
typedef __attribute__((ext_vector_type(8)))  float        v8f;
typedef __attribute__((ext_vector_type(4)))  unsigned int v4u;
typedef __attribute__((ext_vector_type(8)))  int          v8i;
typedef __attribute__((ext_vector_type(4)))  int          v4i;

#define NTOK 98
#define NPAD 112
#define NT   7        // 112/16 row tiles
#define C_   128
#define HEADS_ 4
#define HD   32
// padded LDS strides (elements)
#define XS 136
#define QS 136
#define KS 136
#define VT 136
#define SS 116
#define AS 136
#define OS 136

// ---- WMMA fragment helpers (CDNA5 16x16x32 bf16 layouts) ----
// A (16x32, MxK): lane<16 -> row=lane, K=[kb..kb+7] & [kb+16..kb+23]
//                 lane>=16 -> row=lane-16, K=[kb+8..kb+15] & [kb+24..kb+31]
// B (32x16, KxN): lane picks column N=lane&15, same K split. For C = A*W^T,
// a B column equals a row of W, so both loads are contiguous row reads.
__device__ __forceinline__ v16bf ld_frag16(const __bf16* row_base, int kbase, int hi8) {
  const __bf16* p = row_base + kbase + hi8;
  v16bf r;
#pragma unroll
  for (int i = 0; i < 8; ++i) { r[i] = p[i]; r[i + 8] = p[i + 16]; }
  return r;
}
__device__ __forceinline__ v8f wmma_bf16(v16bf a, v16bf b, v8f c) {
  return __builtin_amdgcn_wmma_f32_16x16x32_bf16(false, a, false, b, (short)0, c, false, false);
}

// ---- Prep kernel: run once per launch; all output identical across blocks of
// the main kernel, so hoist it out. Weights -> bf16 (halves weight traffic,
// kills per-block cvt chains); rel_index/bias_table double-gather -> dense
// per-head bias matrix [H][98][98] (single linear L2-resident load later).
__global__ void win_attn3d_prep(const float* __restrict__ qkv_w,
                                const float* __restrict__ proj_w,
                                const float* __restrict__ bias_table,
                                const int*   __restrict__ rel_index,
                                __bf16* __restrict__ wqkv,
                                __bf16* __restrict__ wproj,
                                float*  __restrict__ relbias) {
  int i = blockIdx.x * blockDim.x + threadIdx.x;
  if (i < 3 * C_ * C_) wqkv[i] = (__bf16)qkv_w[i];
  if (i < C_ * C_)     wproj[i] = (__bf16)proj_w[i];
  if (i < HEADS_ * NTOK * NTOK) {
    int h = i / (NTOK * NTOK), nm = i % (NTOK * NTOK);
    relbias[i] = bias_table[rel_index[nm] * HEADS_ + h];
  }
}

__global__ void win_attn3d_kernel(const float* __restrict__ x,
                                  const float* __restrict__ mask,
                                  const __bf16* __restrict__ wqkv,
                                  const float* __restrict__ qkv_b,
                                  const __bf16* __restrict__ wproj,
                                  const float* __restrict__ proj_b,
                                  const float* __restrict__ relbias,
                                  float* __restrict__ out) {
  extern __shared__ __align__(16) char smem_raw[];
  __bf16* xs   = (__bf16*)smem_raw;          // [NPAD][XS]  staged x (bf16)
  __bf16* qs   = xs   + NPAD * XS;           // [NPAD][QS]  q (pre-scaled)
  __bf16* ks   = qs   + NPAD * QS;           // [NPAD][KS]  k
  __bf16* vt   = ks   + NPAD * KS;           // [C_][VT]    v transposed (chan-major)
  __bf16* attn = vt   + C_   * VT;           // [NPAD][AS]  softmax probs (bf16)
  __bf16* outs = attn + NPAD * AS;           // [NPAD][OS]  attn@v result
  float*  score = (float*)(outs + NPAD * OS);// [NPAD][SS]  raw scores (f32)
  float*  xstage = score;                    // alias: f32 x staging (TDM dest),
                                             // 98*128*4 = 50176B <= 112*116*4

  const int b       = blockIdx.x;
  const int tid     = threadIdx.x;
  const int wv      = tid >> 5;
  const int lane    = tid & 31;
  const int lane_lo = lane & 15;             // D col / fragment row-in-tile
  const int hi8     = (lane >> 4) << 3;      // 8 if upper half-wave
  const float scl   = 0.17677669529663687f;  // 1/sqrt(32)
  const int w_idx   = b & 63;                // window index (B//nW, nW) split

  const float* xb = x + (size_t)b * NTOK * C_;

  // ---- Phase 1: TDM-stage x[b] (98x128 f32 2D tile) into LDS, overlap with
  // zeroing v's token padding; then convert to bf16 with row padding.
#if __has_builtin(__builtin_amdgcn_tensor_load_to_lds) && __has_builtin(__builtin_amdgcn_s_wait_tensorcnt)
  if (wv == 0) {
    // D# group 0: count=1 | lds_addr | global_addr(57b) | type=2
    unsigned lds_addr = (unsigned)(uintptr_t)(void*)xstage; // low 32b == LDS offset
    unsigned long long ga = (unsigned long long)(uintptr_t)xb;
    v4u g0;
    g0[0] = 1u;
    g0[1] = lds_addr;
    g0[2] = (unsigned)(ga & 0xFFFFFFFFu);
    g0[3] = (unsigned)((ga >> 32) & 0x01FFFFFFu) | (2u << 30);
    // D# group 1: data_size=2 (4B); tensor 128x98, stride0=128; tile 128x98
    v8i g1 = {};
    g1[0] = (2 << 16);                                  // data_size
    g1[1] = (int)((C_ & 0xFFFF) << 16);                 // tensor_dim0[15:0]
    g1[2] = (int)(((C_ >> 16) & 0xFFFF) | ((NTOK & 0xFFFF) << 16)); // dim0 hi | dim1 lo
    g1[3] = (int)(((NTOK >> 16) & 0xFFFF) | (C_ << 16));            // dim1 hi | tile_dim0
    g1[4] = NTOK;                                       // tile_dim1 (tile_dim2=0)
    g1[5] = C_;                                         // tensor_dim0_stride[31:0]
    g1[6] = 0; g1[7] = 0;
    v4i g2 = {}; v4i g3 = {};
    v8i g4 = {};   // 6-arg toolchain form: extra group, unused for 2D tile
    __builtin_amdgcn_tensor_load_to_lds(g0, g1, g2, g3, g4, 0);
  }
  for (int i = tid; i < C_ * (VT - NTOK); i += 256) {   // zero v token padding
    int r = i / (VT - NTOK), c = NTOK + i % (VT - NTOK);
    vt[r * VT + c] = (__bf16)0.f;
  }
  if (wv == 0) __builtin_amdgcn_s_wait_tensorcnt(0);
  __syncthreads();
  for (int i = tid; i < NPAD * C_; i += 256) {
    int r = i >> 7, c = i & 127;
    xs[r * XS + c] = (__bf16)((r < NTOK) ? xstage[r * C_ + c] : 0.f);
  }
#else
  for (int i = tid; i < C_ * (VT - NTOK); i += 256) {
    int r = i / (VT - NTOK), c = NTOK + i % (VT - NTOK);
    vt[r * VT + c] = (__bf16)0.f;
  }
  for (int i = tid; i < NPAD * C_; i += 256) {
    int r = i >> 7, c = i & 127;
    xs[r * XS + c] = (__bf16)((r < NTOK) ? xb[r * C_ + c] : 0.f);
  }
#endif
  __syncthreads();

  // ---- Phase 2: QKV GEMM  (7 row-tiles x 24 col-tiles over 3C=384)
  for (int job = wv; job < NT * 24; job += 8) {
    int rt = job % NT, ct = job / NT;
    int cglob = ct * 16;
    const __bf16* wrow = wqkv + (size_t)(cglob + lane_lo) * C_;
    const __bf16* arow = xs + (rt * 16 + lane_lo) * XS;
    v8f acc = {};
#pragma unroll
    for (int kk = 0; kk < C_; kk += 32)
      acc = wmma_bf16(ld_frag16(arow, kk, hi8), ld_frag16(wrow, kk, hi8), acc);
    int comp = cglob >> 7;        // 0=q 1=k 2=v
    int c0   = cglob & 127;       // = h*32 + d0
    float bias = qkv_b[cglob + lane_lo];
    if (comp == 0) {
#pragma unroll
      for (int j = 0; j < 8; ++j)
        qs[(rt * 16 + j + hi8) * QS + c0 + lane_lo] = (__bf16)((acc[j] + bias) * scl);
    } else if (comp == 1) {
#pragma unroll
      for (int j = 0; j < 8; ++j)
        ks[(rt * 16 + j + hi8) * KS + c0 + lane_lo] = (__bf16)(acc[j] + bias);
    } else {
#pragma unroll
      for (int j = 0; j < 8; ++j)
        vt[(c0 + lane_lo) * VT + rt * 16 + j + hi8] = (__bf16)(acc[j] + bias);
    }
  }
  __syncthreads();

  // ---- Phase 3: per-head attention
  const float* maskw = mask + (size_t)w_idx * NTOK * NTOK;
  for (int h = 0; h < HEADS_; ++h) {
    const float* rb = relbias + (size_t)h * NTOK * NTOK;
    // scores: one WMMA per 16x16 tile (hd == K == 32)
    for (int job = wv; job < NT * NT; job += 8) {
      int rt = job % NT, ctm = job / NT;
      v16bf a  = ld_frag16(qs + (rt  * 16 + lane_lo) * QS, h * HD, hi8);
      v16bf bb = ld_frag16(ks + (ctm * 16 + lane_lo) * KS, h * HD, hi8);
      v8f acc = {};
      acc = wmma_bf16(a, bb, acc);
      int mcol = ctm * 16 + lane_lo;
      if (mcol < NTOK) {
#pragma unroll
        for (int j = 0; j < 8; ++j) {
          int nrow = rt * 16 + j + hi8;
          int nr = (nrow < NTOK) ? nrow : (NTOK - 1);
          score[nrow * SS + mcol] = acc[j] + rb[nr * NTOK + mcol] + maskw[nr * NTOK + mcol];
        }
      }
    }
    __syncthreads();
    // softmax: one wave per row, shfl reductions over 98 cols
    for (int row = wv; row < NPAD; row += 8) {
      const float* srow = score + row * SS;
      float vals[4], m = -3.0e38f;
#pragma unroll
      for (int k = 0; k < 4; ++k) {
        int i = lane + 32 * k;
        vals[k] = (i < NTOK) ? srow[i] : -3.0e38f;
        m = fmaxf(m, vals[k]);
      }
#pragma unroll
      for (int o = 16; o > 0; o >>= 1) m = fmaxf(m, __shfl_xor(m, o, 32));
      float s = 0.f;
#pragma unroll
      for (int k = 0; k < 4; ++k) {
        vals[k] = (lane + 32 * k < NTOK) ? __expf(vals[k] - m) : 0.f;
        s += vals[k];
      }
#pragma unroll
      for (int o = 16; o > 0; o >>= 1) s += __shfl_xor(s, o, 32);
      float inv = 1.f / s;
      __bf16* arow = attn + row * AS;
#pragma unroll
      for (int k = 0; k < 4; ++k)            // covers cols 0..127 (>=98 -> 0)
        arow[lane + 32 * k] = (__bf16)(vals[k] * inv);
    }
    __syncthreads();
    // attn @ v  (7 x 2 tiles, K = 128 padded tokens)
    for (int job = wv; job < NT * 2; job += 8) {
      int rt = job % NT, ct = job / NT;
      const __bf16* arow = attn + (rt * 16 + lane_lo) * AS;
      const __bf16* vrow = vt + (h * HD + ct * 16 + lane_lo) * VT;
      v8f acc = {};
#pragma unroll
      for (int kk = 0; kk < 128; kk += 32)
        acc = wmma_bf16(ld_frag16(arow, kk, hi8), ld_frag16(vrow, kk, hi8), acc);
#pragma unroll
      for (int j = 0; j < 8; ++j)
        outs[(rt * 16 + j + hi8) * OS + h * HD + ct * 16 + lane_lo] = (__bf16)acc[j];
    }
    __syncthreads();
  }

  // ---- Phase 4: projection GEMM + bias -> global
  float* ob = out + (size_t)b * NTOK * C_;
  for (int job = wv; job < NT * 8; job += 8) {
    int rt = job % NT, ct = job / NT;
    const __bf16* arow = outs + (rt * 16 + lane_lo) * OS;
    const __bf16* wrow = wproj + (size_t)(ct * 16 + lane_lo) * C_;
    v8f acc = {};
#pragma unroll
    for (int kk = 0; kk < C_; kk += 32)
      acc = wmma_bf16(ld_frag16(arow, kk, hi8), ld_frag16(wrow, kk, hi8), acc);
    float bias = proj_b[ct * 16 + lane_lo];
#pragma unroll
    for (int j = 0; j < 8; ++j) {
      int nrow = rt * 16 + j + hi8;
      if (nrow < NTOK) ob[nrow * C_ + ct * 16 + lane_lo] = acc[j] + bias;
    }
  }
}

extern "C" void kernel_launch(void* const* d_in, const int* in_sizes, int n_in,
                              void* d_out, int out_size, void* d_ws, size_t ws_size,
                              hipStream_t stream) {
  (void)in_sizes; (void)n_in; (void)out_size; (void)ws_size;
  const float* x          = (const float*)d_in[0];
  const float* mask       = (const float*)d_in[1];
  const float* qkv_w      = (const float*)d_in[2];
  const float* qkv_b      = (const float*)d_in[3];
  const float* proj_w     = (const float*)d_in[4];
  const float* proj_b     = (const float*)d_in[5];
  const float* bias_table = (const float*)d_in[6];
  const int*   rel_index  = (const int*)d_in[7];
  float* o = (float*)d_out;

  // workspace: bf16 qkv_w | bf16 proj_w | f32 relbias[H][98][98]  (~285 KB)
  __bf16* wqkv   = (__bf16*)d_ws;
  __bf16* wproj  = wqkv + 3 * C_ * C_;
  float*  relbias = (float*)(wproj + C_ * C_);

  win_attn3d_prep<<<192, 256, 0, stream>>>(qkv_w, proj_w, bias_table, rel_index,
                                           wqkv, wproj, relbias);

  size_t smem = (size_t)(NPAD * XS + NPAD * QS + NPAD * KS + C_ * VT +
                         NPAD * AS + NPAD * OS) * sizeof(__bf16) +
                (size_t)NPAD * SS * sizeof(float);   // ~233.5 KB (< 320 KB/WGP)

  win_attn3d_kernel<<<2048, 256, smem, stream>>>(
      x, mask, wqkv, qkv_b, wproj, proj_b, relbias, o);
}
